// TacotronDecoder_77103252898099
// MI455X (gfx1250) — matrix operand order repaired and checked by reading, not compiled, not run on hardware
//
#include <hip/hip_runtime.h>

// ---------------------------------------------------------------------------
// CDNA5 (gfx1250) Tacotron decoder — all GEMMs on v_wmma_f32_16x16x32_bf16.
//   - Weights packed fp32->bf16 once per call in a contiguous-k B-fragment layout
//     (lane n=L%16, k=(L/16)*16+e): one contiguous 32B load per lane/k-step.
//   - ALL activations consumed by GEMMs are kept in bf16 (produced at source:
//     LSTM act, softmax ctx, prenet, im2col). In the same contiguous-k order a
//     lane's fragment is two contiguous 16B runs of row m=L%16, so the GEMM
//     A path is two global_load_b128 per k-step: no LDS, no barriers, no
//     conversion VALU on the recurrence critical path.
//   - Inner loop is branch-free (second n-tile clamped, predicated at store
//     only) -> no per-iteration EXEC save/restore around the WMMAs.
//   - Recurrent weights (~36 MB bf16) and postnet col buffer are L2-resident
//     on MI455X (192 MB L2), so the T=192 recurrence never touches HBM.
// ---------------------------------------------------------------------------

typedef __attribute__((ext_vector_type(16))) __bf16  v16bf;
typedef __attribute__((ext_vector_type(8)))  float   v8f;

union AFrag { uint4 q[2]; v16bf b; };

__device__ __forceinline__ unsigned short f2bf(float f) {
  union { float f; unsigned int u; } x; x.f = f;
  unsigned int u = x.u;
  u += 0x7FFFu + ((u >> 16) & 1u);   // round-to-nearest-even
  return (unsigned short)(u >> 16);
}

__device__ __forceinline__ float bf2f(unsigned short s) {
  union { unsigned int u; float f; } x; x.u = (unsigned int)s << 16;
  return x.f;
}

struct Seg {
  const unsigned short* A;   // bf16 activations, row stride lda (lda%32==0),
                             // columns zero-padded to ceil(K/32)*32
  const unsigned short* W;   // packed bf16 weights [ceil(K/32)][N/16][32][16]
  int lda;
  int K;
};

// ---------------------------------------------------------------------------
// Weight pack: fp32 [K,N] -> bf16 in WMMA B-fragment order.
// ---------------------------------------------------------------------------
__global__ void pack_w(const float* __restrict__ W, unsigned short* __restrict__ Wpk,
                       int K, int N) {
  int Nt = N >> 4;
  int Kt = (K + 31) >> 5;
  size_t total = (size_t)Kt * Nt * 512;
  for (size_t idx = (size_t)blockIdx.x * blockDim.x + threadIdx.x; idx < total;
       idx += (size_t)gridDim.x * blockDim.x) {
    int e    = (int)(idx & 15);
    int lane = (int)((idx >> 4) & 31);
    size_t t = idx >> 9;
    int nt = (int)(t % Nt);
    int kt = (int)(t / Nt);
    int n = nt * 16 + (lane & 15);
    int k = kt * 32 + (lane >> 4) * 16 + e;
    float v = (k < K) ? W[(size_t)k * N + n] : 0.f;
    Wpk[idx] = f2bf(v);
  }
}

// fp32 [R,C] -> bf16 [R,Cp], zero-padded columns C..Cp.
__global__ void to_bf16(const float* __restrict__ x, unsigned short* __restrict__ y,
                        int R, int C, int Cp) {
  size_t total = (size_t)R * Cp;
  for (size_t i = (size_t)blockIdx.x * blockDim.x + threadIdx.x; i < total;
       i += (size_t)gridDim.x * blockDim.x) {
    int c = (int)(i % Cp);
    size_t r = i / Cp;
    y[i] = (c < C) ? f2bf(x[r * C + c]) : (unsigned short)0;
  }
}

// ---------------------------------------------------------------------------
// M=16-row WMMA GEMM: C = sum(segs: A_i @ W_i) (+bias)(+relu)
// grid = (ceil(Ntiles/8), Mrows/16); block = 128 (4 independent waves).
// Each wave owns two consecutive n-tiles sharing one A fragment; the second
// tile index is clamped so the inner loop is branch-free (predicate applied
// only at the store).
// Inner loop: 2x global_load_b128 (A runs) + 4x global_load_b128 (B frags)
// + 2x v_wmma_f32_16x16x32_bf16. No LDS, no barriers, no EXEC juggling.
// flags: bit0 = relu, bit1 = bf16 output.
// permMode: 0 identity, 1: r=b*pT+t -> t*pB+b, 2: r=t*pB+b -> b*pT+t.
// ---------------------------------------------------------------------------
__global__ void __launch_bounds__(128)
gemm16_wmma(Seg s0, Seg s1, Seg s2,
            const float* __restrict__ bias, void* __restrict__ Cout, int ldc,
            int Ntiles, int flags, int permMode, int pT, int pB) {
  const int lane = threadIdx.x & 31;
  const int wave = threadIdx.x >> 5;
  const int nt0  = (blockIdx.x * 4 + wave) * 2;
  if (nt0 >= Ntiles) return;                  // wave-uniform, no barriers used
  const bool two = (nt0 + 1 < Ntiles);
  const int nt1  = two ? nt0 + 1 : nt0;       // clamp: loads always safe
  const int m0   = blockIdx.y * 16;
  const int m    = lane & 15;
  const int half = lane >> 4;

  v8f acc0 = {0.f, 0.f, 0.f, 0.f, 0.f, 0.f, 0.f, 0.f};
  v8f acc1 = {0.f, 0.f, 0.f, 0.f, 0.f, 0.f, 0.f, 0.f};

  Seg segs[3] = {s0, s1, s2};
  for (int si = 0; si < 3; ++si) {
    Seg sg = segs[si];
    if (sg.A == nullptr) continue;
    const int nkt = (sg.K + 31) >> 5;
    const v16bf* wp = reinterpret_cast<const v16bf*>(sg.W);
    // lane's A row, as 16B units (lda%32==0 -> rows 16B aligned)
    const uint4* ap = reinterpret_cast<const uint4*>(
        sg.A + (size_t)(m0 + m) * sg.lda);
    for (int kt = 0; kt < nkt; ++kt) {
      AFrag af;                                // contiguous-k A layout (as B pack): two runs
      af.q[0] = ap[kt * 4 + 2 * half];         // k = kt*32 + half*16 .. +8
      af.q[1] = ap[kt * 4 + 2 * half + 1];     // k = kt*32 + half*16 + 8 .. +8
      v16bf b0 = wp[((size_t)kt * Ntiles + nt0) * 32 + lane];
      v16bf b1 = wp[((size_t)kt * Ntiles + nt1) * 32 + lane];
      acc0 = __builtin_amdgcn_wmma_f32_16x16x32_bf16(
          false, af.b, false, b0, (short)0, acc0, false, false);
      acc1 = __builtin_amdgcn_wmma_f32_16x16x32_bf16(
          false, af.b, false, b1, (short)0, acc1, false, false);
    }
  }
  // C/D layout: lane n=L%16; VGPR r -> m = (L/16)*8 + r.
  int nl = lane & 15;
  int mb = (lane >> 4) * 8;
#pragma unroll
  for (int h2 = 0; h2 < 2; ++h2) {
    if (h2 && !two) break;
    int nt = nt0 + h2;
    int n = nt * 16 + nl;
    v8f acc = h2 ? acc1 : acc0;
#pragma unroll
    for (int r = 0; r < 8; ++r) {
      int mm = m0 + mb + r;
      float v = acc[r];
      if (bias) v += bias[n];
      if (flags & 1) v = fmaxf(v, 0.f);
      size_t orow;
      if (permMode == 1)      orow = (size_t)(mm % pT) * pB + (size_t)(mm / pT);
      else if (permMode == 2) orow = (size_t)(mm % pB) * pT + (size_t)(mm / pB);
      else                    orow = (size_t)mm;
      if (flags & 2) ((unsigned short*)Cout)[orow * (size_t)ldc + n] = f2bf(v);
      else           ((float*)Cout)[orow * (size_t)ldc + n] = v;
    }
  }
}

// ---------------------------------------------------------------------------
// LSTM gate activations (gate order i,f,g,o); h stored as bf16 (GEMM operand),
// c stays fp32.
// ---------------------------------------------------------------------------
__global__ void lstm_act(const float* __restrict__ z, unsigned short* __restrict__ h,
                         float* __restrict__ c, int Adim,
                         unsigned short* __restrict__ outH, int ldOut) {
  int idx = blockIdx.x * blockDim.x + threadIdx.x;
  if (idx >= 16 * Adim) return;
  int b = idx / Adim, j = idx - b * Adim;
  const float* zb = z + (size_t)b * 4 * Adim;
  float gi = zb[j], gf = zb[Adim + j], gg = zb[2 * Adim + j], go = zb[3 * Adim + j];
  float si = 1.f / (1.f + __expf(-gi));
  float sf = 1.f / (1.f + __expf(-gf));
  float so = 1.f / (1.f + __expf(-go));
  float c2 = sf * c[idx] + si * tanhf(gg);
  float h2 = so * tanhf(c2);
  c[idx] = c2;
  unsigned short hb = f2bf(h2);
  h[idx] = hb;
  if (outH) outH[(size_t)b * ldOut + j] = hb;
}

// Record pre-update attention weights: attn_out[b, t, s] = aw[b, s]
__global__ void copy_aw(const float* __restrict__ aw, float* __restrict__ outA, int t) {
  int idx = blockIdx.x * blockDim.x + threadIdx.x;
  if (idx >= 16 * 384) return;
  int b = idx / 384, s = idx - b * 384;
  outA[((size_t)b * 192 + t) * 384 + s] = aw[idx];
}

// Location conv: [B,S,2] (*) [31,2,32] 'same' -> lconv bf16 [B,S,32]
__global__ void loc_conv31(const float* __restrict__ aw, const float* __restrict__ awc,
                           const float* __restrict__ kc,
                           unsigned short* __restrict__ lc) {
  int idx = blockIdx.x * blockDim.x + threadIdx.x;
  if (idx >= 16 * 384 * 32) return;
  int f = idx & 31;
  int bs = idx >> 5;
  int s = bs % 384, b = bs / 384;
  float acc = 0.f;
#pragma unroll 1
  for (int k = 0; k < 31; ++k) {
    int sp = s + k - 15;
    if (sp < 0 || sp >= 384) continue;
    acc += aw[b * 384 + sp] * kc[k * 64 + f] + awc[b * 384 + sp] * kc[k * 64 + 32 + f];
  }
  lc[idx] = f2bf(acc);
}

// e[b,s] = sum_d tanh(q[b,d] + loc[b,s,d] + pm[b,s,d]) * v[d]
__global__ void energies(const float* __restrict__ q, const float* __restrict__ locAD,
                         const float* __restrict__ pm, const float* __restrict__ vv,
                         float* __restrict__ e) {
  int idx = blockIdx.x * blockDim.x + threadIdx.x;
  if (idx >= 16 * 384) return;
  int b = idx / 384;
  const float* qb = q + b * 128;
  const float* l  = locAD + (size_t)idx * 128;
  const float* p  = pm + (size_t)idx * 128;
  float acc = 0.f;
  for (int d = 0; d < 128; ++d) acc += tanhf(qb[d] + l[d] + p[d]) * vv[d];
  e[idx] = acc;
}

// softmax over S, aw/awc update, context = aw2 @ memory (bf16 out for GEMMs).
__global__ void __launch_bounds__(512)
softmax_ctx(const float* __restrict__ e, float* __restrict__ aw, float* __restrict__ awc,
            const float* __restrict__ memory, unsigned short* __restrict__ ctx,
            unsigned short* __restrict__ outsRow) {
  int b = blockIdx.x;
  __shared__ float sp[384];
  __shared__ float red[512];
  int tid = threadIdx.x;
  float v = (tid < 384) ? e[b * 384 + tid] : -1e30f;
  red[tid] = v; __syncthreads();
  for (int s = 256; s > 0; s >>= 1) {
    if (tid < s) red[tid] = fmaxf(red[tid], red[tid + s]);
    __syncthreads();
  }
  float mx = red[0]; __syncthreads();
  float ex = (tid < 384) ? __expf(v - mx) : 0.f;
  red[tid] = ex; __syncthreads();
  for (int s = 256; s > 0; s >>= 1) {
    if (tid < s) red[tid] += red[tid + s];
    __syncthreads();
  }
  float inv = 1.f / red[0];
  if (tid < 384) {
    float p = ex * inv;
    sp[tid] = p;
    aw[b * 384 + tid] = p;
    awc[b * 384 + tid] += p;
  }
  __syncthreads();
  float acc = 0.f;  // tid = e-channel (E=512); coalesced over memory
  for (int s = 0; s < 384; ++s)
    acc += sp[s] * memory[((size_t)b * 384 + s) * 512 + tid];
  unsigned short cb = f2bf(acc);
  ctx[b * 512 + tid] = cb;
  outsRow[(size_t)b * 1536 + 1024 + tid] = cb;
}

// gate[b,t] = sigmoid(outs_row . gate_W + gate_b); outs rows (t,b), bf16.
__global__ void gate_out(const unsigned short* __restrict__ outs,
                         const float* __restrict__ gW,
                         const float* __restrict__ gb, float* __restrict__ g) {
  int r = blockIdx.x * blockDim.x + threadIdx.x;
  if (r >= 3072) return;
  const unsigned short* row = outs + (size_t)r * 1536;
  float acc = gb[0];
  for (int k = 0; k < 1536; ++k) acc += bf2f(row[k]) * gW[k];
  int t = r >> 4, b = r & 15;
  g[b * 192 + t] = 1.f / (1.f + __expf(-acc));
}

// im2col for kernel-5 'same' conv: x fp32 [B,T,Ci] rows (b,t) -> bf16 col
// [3072, Cp] with zero pad in columns 5*Ci..Cp.
__global__ void im2col5(const float* __restrict__ x, int Ci,
                        unsigned short* __restrict__ col, int Cp) {
  size_t total = (size_t)3072 * Cp;
  int W5 = 5 * Ci;
  for (size_t i = (size_t)blockIdx.x * blockDim.x + threadIdx.x; i < total;
       i += (size_t)gridDim.x * blockDim.x) {
    int cc = (int)(i % Cp);
    size_t r = i / Cp;
    unsigned short v = 0;
    if (cc < W5) {
      int tap = cc / Ci, c = cc - tap * Ci;
      int t = (int)(r % 192), b = (int)(r / 192);
      int tp = t + tap - 2;
      if (tp >= 0 && tp < 192) v = f2bf(x[((size_t)b * 192 + tp) * Ci + c]);
    }
    col[i] = v;
  }
}

// optional tanh then inference batchnorm, in place (fp32).
__global__ void bn_act(float* __restrict__ y, int Co, const float* __restrict__ gm,
                       const float* __restrict__ bt, const float* __restrict__ mn,
                       const float* __restrict__ vr, int do_tanh) {
  size_t i = (size_t)blockIdx.x * blockDim.x + threadIdx.x;
  if (i >= (size_t)3072 * Co) return;
  int c = (int)(i % Co);
  float v = y[i];
  if (do_tanh) v = tanhf(v);
  v = (v - mn[c]) * rsqrtf(vr[c] + 1e-5f) * gm[c] + bt[c];
  y[i] = v;
}

__global__ void add_res(const float* __restrict__ a, const float* __restrict__ b,
                        float* __restrict__ o, int n) {
  int i = blockIdx.x * blockDim.x + threadIdx.x;
  if (i < n) o[i] = a[i] + b[i];
}

__global__ void zero_f32(float* p, int n) {
  int i = blockIdx.x * blockDim.x + threadIdx.x;
  if (i < n) p[i] = 0.f;
}

__global__ void zero_u16(unsigned short* p, int n) {
  int i = blockIdx.x * blockDim.x + threadIdx.x;
  if (i < n) p[i] = 0;
}

// ---------------------------------------------------------------------------
extern "C" void kernel_launch(void* const* d_in, const int* in_sizes, int n_in,
                              void* d_out, int out_size, void* d_ws, size_t ws_size,
                              hipStream_t stream) {
  (void)in_sizes; (void)n_in; (void)out_size; (void)ws_size;
  const int B = 16, S = 384, E = 512, T = 192, M = 80, P = 256;
  const int A = 1024, D = 1024, AD = 128;

  const float* memory    = (const float*)d_in[0];
  const float* mel_in    = (const float*)d_in[1];
  const float* prenet_w1 = (const float*)d_in[2];
  const float* prenet_w2 = (const float*)d_in[3];
  const float* attn_Wx   = (const float*)d_in[4];
  const float* attn_Wh   = (const float*)d_in[5];
  const float* attn_b    = (const float*)d_in[6];
  const float* q_W       = (const float*)d_in[7];
  const float* mem_W     = (const float*)d_in[8];
  const float* loc_k     = (const float*)d_in[9];
  const float* loc_W     = (const float*)d_in[10];
  const float* attn_v    = (const float*)d_in[11];
  const float* dec_Wx    = (const float*)d_in[12];
  const float* dec_Wh    = (const float*)d_in[13];
  const float* dec_b     = (const float*)d_in[14];
  const float* proj_W    = (const float*)d_in[15];
  const float* proj_b    = (const float*)d_in[16];
  const float* gate_W    = (const float*)d_in[17];
  const float* gate_b    = (const float*)d_in[18];
  const float* post_k[5]; const float* post_bs[5];
  const float* bn_g[5]; const float* bn_b[5]; const float* bn_m[5]; const float* bn_v[5];
  for (int i = 0; i < 5; ++i) {
    post_k[i]  = (const float*)d_in[19 + i];
    post_bs[i] = (const float*)d_in[24 + i];
    bn_g[i]    = (const float*)d_in[29 + i];
    bn_b[i]    = (const float*)d_in[34 + i];
    bn_m[i]    = (const float*)d_in[39 + i];
    bn_v[i]    = (const float*)d_in[44 + i];
  }

  float* out_mel  = (float*)d_out;                 // [B,T,80]
  float* out_post = out_mel + (size_t)B * T * M;   // [B,T,80]
  float* out_gate = out_post + (size_t)B * T * M;  // [B,T]
  float* out_attn = out_gate + (size_t)B * T;      // [B,T,S]

  // ---- workspace bump allocator ----
  size_t off = 0;
  auto allocB = [&](size_t bytes) -> char* {
    char* p = (char*)d_ws + off;
    off += (bytes + 255) & ~(size_t)255;
    return p;
  };
  auto PK = [&](const float* W, int K, int N) -> unsigned short* {
    size_t elems = (size_t)((K + 31) / 32) * 32 * N;
    unsigned short* p = (unsigned short*)allocB(elems * 2);
    pack_w<<<2048, 256, 0, stream>>>(W, p, K, N);
    return p;
  };

  // bf16-packed weights (L2-resident working set on MI455X)
  unsigned short* pw1  = PK(prenet_w1, M, P);
  unsigned short* pw2  = PK(prenet_w2, P, P);
  unsigned short* wxx  = PK(attn_Wx, P, 4 * A);                        // xt rows
  unsigned short* wxc  = PK(attn_Wx + (size_t)P * 4 * A, E, 4 * A);    // ctx rows
  unsigned short* whh  = PK(attn_Wh, A, 4 * A);
  unsigned short* qwp  = PK(q_W, A, AD);
  unsigned short* mwp  = PK(mem_W, E, AD);
  unsigned short* lwp  = PK(loc_W, 32, AD);
  unsigned short* wdx0 = PK(dec_Wx, A, 4 * D);                         // ah rows
  unsigned short* wdx1 = PK(dec_Wx + (size_t)A * 4 * D, E, 4 * D);     // ctx rows
  unsigned short* wdh  = PK(dec_Wh, D, 4 * D);
  unsigned short* pjw  = PK(proj_W, D + E, M);
  int pKs[5] = {5 * M, 5 * 512, 5 * 512, 5 * 512, 5 * 512};
  int pNs[5] = {512, 512, 512, 512, M};
  int pCp[5] = {416, 2560, 2560, 2560, 2560};   // padded im2col widths
  unsigned short* ppk[5];
  for (int i = 0; i < 5; ++i) ppk[i] = PK(post_k[i], pKs[i], pNs[i]);

  // bf16 activation buffers (GEMM A operands)
  unsigned short* mem_bf = (unsigned short*)allocB((size_t)B * S * E * 2);
  unsigned short* mel_bf = (unsigned short*)allocB((size_t)B * T * 96 * 2);
  unsigned short* ptmp   = (unsigned short*)allocB((size_t)B * T * P * 2);
  unsigned short* dec_tb = (unsigned short*)allocB((size_t)T * B * P * 2);
  unsigned short* lconv  = (unsigned short*)allocB((size_t)B * S * 32 * 2);
  unsigned short* ah_bf  = (unsigned short*)allocB((size_t)B * A * 2);
  unsigned short* dh_bf  = (unsigned short*)allocB((size_t)B * D * 2);
  unsigned short* ctx_bf = (unsigned short*)allocB((size_t)B * E * 2);
  unsigned short* outs   = (unsigned short*)allocB((size_t)T * B * (D + E) * 2);
  unsigned short* col    = (unsigned short*)allocB((size_t)B * T * 2560 * 2);
  // fp32 buffers
  float* pm    = (float*)allocB((size_t)B * S * AD * 4);
  float* zbuf  = (float*)allocB((size_t)B * 4 * A * 4);
  float* qbuf  = (float*)allocB((size_t)B * AD * 4);
  float* locAD = (float*)allocB((size_t)B * S * AD * 4);
  float* ebuf  = (float*)allocB((size_t)B * S * 4);
  float* ac    = (float*)allocB((size_t)B * A * 4);
  float* dc    = (float*)allocB((size_t)B * D * 4);
  float* aw    = (float*)allocB((size_t)B * S * 4);
  float* awc   = (float*)allocB((size_t)B * S * 4);
  float* pbufA = (float*)allocB((size_t)B * T * 512 * 4);
  float* pbufB = (float*)allocB((size_t)B * T * 512 * 4);

  Seg nil = {nullptr, nullptr, 0, 0};
  auto GEMM = [&](Seg a, Seg b2, Seg c2, const float* bias, void* C, int ldc,
                  int Mrows, int N, int flags, int perm, int pTv, int pBv) {
    int Ntiles = N / 16;
    dim3 grid((Ntiles + 7) / 8, Mrows / 16);
    gemm16_wmma<<<grid, 128, 0, stream>>>(a, b2, c2, bias, C, ldc, Ntiles,
                                          flags, perm, pTv, pBv);
  };

  // ---- zero initial state ----
  zero_u16<<<(B * A + 255) / 256, 256, 0, stream>>>(ah_bf, B * A);
  zero_u16<<<(B * D + 255) / 256, 256, 0, stream>>>(dh_bf, B * D);
  zero_u16<<<(B * E + 255) / 256, 256, 0, stream>>>(ctx_bf, B * E);
  zero_f32<<<(B * A + 255) / 256, 256, 0, stream>>>(ac, B * A);
  zero_f32<<<(B * D + 255) / 256, 256, 0, stream>>>(dc, B * D);
  zero_f32<<<(B * S + 255) / 256, 256, 0, stream>>>(aw, B * S);
  zero_f32<<<(B * S + 255) / 256, 256, 0, stream>>>(awc, B * S);

  // ---- bf16 copies of fp32 inputs used as GEMM A operands ----
  to_bf16<<<2048, 256, 0, stream>>>(memory, mem_bf, B * S, E, E);
  to_bf16<<<2048, 256, 0, stream>>>(mel_in, mel_bf, B * T, M, 96);

  // ---- parallel pre-pass ----
  // processed_memory = memory @ mem_W : [6144,512]x[512,128]
  GEMM({mem_bf, mwp, E, E}, nil, nil, nullptr, pm, AD, B * S, AD, 0, 0, 0, 0);
  // prenet: relu(relu(mel @ w1) @ w2); bf16 out; rows (b,t) -> store (t,b)
  GEMM({mel_bf, pw1, 96, M}, nil, nil, nullptr, ptmp, P, B * T, P, 3, 0, 0, 0);
  GEMM({ptmp, pw2, P, P}, nil, nil, nullptr, dec_tb, P, B * T, P, 3, 1, T, B);

  // ---- sequential decode: T steps ----
  for (int t = 0; t < T; ++t) {
    // attention LSTM gates: [xt | ctx] @ Wx + ah @ Wh + b
    GEMM({dec_tb + (size_t)t * B * P, wxx, P, P},
         {ctx_bf, wxc, E, E},
         {ah_bf, whh, A, A},
         attn_b, zbuf, 4 * A, B, 4 * A, 0, 0, 0, 0);
    lstm_act<<<(B * A + 255) / 256, 256, 0, stream>>>(zbuf, ah_bf, ac, A, nullptr, 0);

    // record pre-update attention weights as this step's output
    copy_aw<<<(B * S + 255) / 256, 256, 0, stream>>>(aw, out_attn, t);

    // location-sensitive attention
    loc_conv31<<<(B * S * 32 + 255) / 256, 256, 0, stream>>>(aw, awc, loc_k, lconv);
    GEMM({lconv, lwp, 32, 32}, nil, nil, nullptr, locAD, AD, B * S, AD, 0, 0, 0, 0);
    GEMM({ah_bf, qwp, A, A}, nil, nil, nullptr, qbuf, AD, B, AD, 0, 0, 0, 0);
    energies<<<(B * S + 255) / 256, 256, 0, stream>>>(qbuf, locAD, pm, attn_v, ebuf);
    softmax_ctx<<<B, 512, 0, stream>>>(ebuf, aw, awc, memory, ctx_bf,
                                       outs + (size_t)t * B * (D + E));

    // decoder LSTM gates: [ah | ctx] @ dec_Wx + dh @ dec_Wh + b
    GEMM({ah_bf, wdx0, A, A},
         {ctx_bf, wdx1, E, E},
         {dh_bf, wdh, D, D},
         dec_b, zbuf, 4 * D, B, 4 * D, 0, 0, 0, 0);
    lstm_act<<<(B * D + 255) / 256, 256, 0, stream>>>(
        zbuf, dh_bf, dc, D, outs + (size_t)t * B * (D + E), D + E);
  }

  // ---- outputs ----
  // mel = outs @ proj_W + proj_b   (rows (t,b) -> store (b,t))
  GEMM({outs, pjw, D + E, D + E}, nil, nil, proj_b, out_mel, M, T * B, M, 0, 2, T, B);
  gate_out<<<(T * B + 255) / 256, 256, 0, stream>>>(outs, gate_W, gate_b, out_gate);

  // ---- postnet: 5x (im2col(bf16) + WMMA GEMM + [tanh] + BN), residual ----
  const float* cur = out_mel;
  int curC = M;
  float* bufs[2] = {pbufA, pbufB};
  for (int i = 0; i < 5; ++i) {
    im2col5<<<2048, 256, 0, stream>>>(cur, curC, col, pCp[i]);
    float* dst = bufs[i & 1];
    GEMM({col, ppk[i], pCp[i], pKs[i]}, nil, nil, post_bs[i], dst, pNs[i],
         B * T, pNs[i], 0, 0, 0, 0);
    int nelem = 3072 * pNs[i];
    bn_act<<<(nelem + 255) / 256, 256, 0, stream>>>(dst, pNs[i], bn_g[i], bn_b[i],
                                                    bn_m[i], bn_v[i], i < 4 ? 1 : 0);
    cur = dst;
    curC = pNs[i];
  }
  add_res<<<(B * T * M + 255) / 256, 256, 0, stream>>>(out_mel, pbufA, out_post,
                                                       B * T * M);
}
